// Decoder_4286377361994
// MI455X (gfx1250) — matrix operand order repaired and checked by reading, not compiled, not run on hardware
//
#include <hip/hip_runtime.h>
#include <hip/hip_bf16.h>

typedef _Float16 half_t;
typedef __attribute__((ext_vector_type(16))) _Float16 v16h;
typedef __attribute__((ext_vector_type(8)))  _Float16 v8h;
typedef __attribute__((ext_vector_type(8)))  float    v8f;

#define SEQ   12
#define BATCH 512
#define NG    16
#define GSZ   32
#define HD    128
#define ED    64
#define KFE   192     // E + H
#define PRE   512
#define BOT   1024
#define MLPD  1024
#define KDH   1152    // H + BOT

#define PFE_STRIDE 200   // 192 + 8  (row stride *2B = 400, multiple of 16)
#define X1_STRIDE  528   // 512 + 16 (*2B = 1056, multiple of 16)
#define MID_STRIDE 1040  // 1024+16  (*2B = 2080, multiple of 16)

// ---------------------------------------------------------------------------
// WMMA helper: D = A(16x32 f16) x B(32x16 f16) + C(f32)
// A layout: lane m=L&15, elems e: K = (e&7) + ((e>>3)<<4) + (L>>4)*8
// B layout: lane n=L&15, elems e: K = (e&7) + ((e>>3)<<4) + (L>>4)*8
// C/D     : lane n=L&15, vgpr r: M = r + (L>>4)*8
// ---------------------------------------------------------------------------
__device__ __forceinline__ v8f wmma_f16(v16h a, v16h b, v8f c) {
    return __builtin_amdgcn_wmma_f32_16x16x32_f16(
        false, a, false, b, (short)0, c, false, false);
}

__device__ __forceinline__ v16h cat8(v8h lo, v8h hi) {
    return __builtin_shufflevector(lo, hi, 0,1,2,3,4,5,6,7,8,9,10,11,12,13,14,15);
}

// ---------------------------------------------------------------------------
// One-time setup kernels
// ---------------------------------------------------------------------------
__global__ void cvt_f32_to_f16(const float* __restrict__ src,
                               half_t* __restrict__ dst, int n) {
    int i = blockIdx.x * blockDim.x + threadIdx.x;
    if (i < n) dst[i] = (half_t)src[i];
}

__global__ void init_state(const float* __restrict__ hh, const float* __restrict__ ch,
                           const float* __restrict__ last_pos,
                           float* __restrict__ h, float* __restrict__ c,
                           float* __restrict__ pos) {
    int i = blockIdx.x * blockDim.x + threadIdx.x;
    if (i < BATCH * HD) { h[i] = hh[i]; c[i] = ch[i]; }
    if (i < BATCH * 2)  pos[i] = last_pos[i];
}

// dec_in[t][row][:] : t==0 -> last_pos_rel @ Wse^T + bse ; t>0 -> gt[t-1] @ Wse^T + bse
__global__ void decin_kernel(const float* __restrict__ last_pos_rel,
                             const float* __restrict__ ptr_rel,
                             const float* __restrict__ Wse, const float* __restrict__ bse,
                             float* __restrict__ dec_in_all) {
    int idx = blockIdx.x * blockDim.x + threadIdx.x;
    if (idx >= SEQ * BATCH * ED) return;
    int e = idx & 63;
    int row = (idx >> 6) & 511;
    int t = idx >> 15;
    const float* src = (t == 0) ? (last_pos_rel + row * 2)
                                : (ptr_rel + (size_t)((t - 1) * BATCH + row) * 2);
    dec_in_all[idx] = src[0] * Wse[e * 2 + 0] + src[1] * Wse[e * 2 + 1] + bse[e];
}

// ---------------------------------------------------------------------------
// Per-step LSTM cell (+ rel_pos, cumulative pos, loss partial). 1 block = 1 row.
// ---------------------------------------------------------------------------
__global__ __launch_bounds__(128) void lstm_kernel(
    const float* __restrict__ dec_in,   // [512][64]
    const float* __restrict__ W_ih,     // [512][64]  gates i,f,g,o
    const float* __restrict__ W_hh,     // [512][128]
    const float* __restrict__ b_ih, const float* __restrict__ b_hh,
    const float* __restrict__ Wpos,     // [2][128]
    const float* __restrict__ bpos,     // [2]
    const float* __restrict__ gt,       // [512][2]
    float* __restrict__ h, float* __restrict__ c,   // [512][128] in/out
    float* __restrict__ pos,            // [512][2] in/out
    half_t* __restrict__ dh,            // [512][1152]  (write cols 0..127)
    float* __restrict__ out_traj,       // [512][2]
    float* __restrict__ lossp)          // [512]
{
    __shared__ float sx[ED];
    __shared__ float sh[HD];
    __shared__ float redx[HD];
    __shared__ float redy[HD];
    const int row = blockIdx.x, tid = threadIdx.x;

    if (tid < ED) sx[tid] = dec_in[row * ED + tid];
    sh[tid] = h[row * HD + tid];
    __syncthreads();

    float gi = b_ih[0 * HD + tid] + b_hh[0 * HD + tid];
    float gf = b_ih[1 * HD + tid] + b_hh[1 * HD + tid];
    float gg = b_ih[2 * HD + tid] + b_hh[2 * HD + tid];
    float go = b_ih[3 * HD + tid] + b_hh[3 * HD + tid];
#pragma unroll 4
    for (int k = 0; k < ED; k++) {
        float x = sx[k];
        gi += x * W_ih[(0 * HD + tid) * ED + k];
        gf += x * W_ih[(1 * HD + tid) * ED + k];
        gg += x * W_ih[(2 * HD + tid) * ED + k];
        go += x * W_ih[(3 * HD + tid) * ED + k];
    }
#pragma unroll 4
    for (int k = 0; k < HD; k++) {
        float hv = sh[k];
        gi += hv * W_hh[(0 * HD + tid) * HD + k];
        gf += hv * W_hh[(1 * HD + tid) * HD + k];
        gg += hv * W_hh[(2 * HD + tid) * HD + k];
        go += hv * W_hh[(3 * HD + tid) * HD + k];
    }
    float cv = c[row * HD + tid];
    float sig_i = 1.f / (1.f + __expf(-gi));
    float sig_f = 1.f / (1.f + __expf(-gf));
    float sig_o = 1.f / (1.f + __expf(-go));
    cv = sig_f * cv + sig_i * tanhf(gg);
    float hv = sig_o * tanhf(cv);
    c[row * HD + tid] = cv;
    h[row * HD + tid] = hv;
    dh[(size_t)row * KDH + tid] = (half_t)hv;

    redx[tid] = hv * Wpos[tid];
    redy[tid] = hv * Wpos[HD + tid];
    __syncthreads();
    for (int s = 64; s > 0; s >>= 1) {
        if (tid < s) { redx[tid] += redx[tid + s]; redy[tid] += redy[tid + s]; }
        __syncthreads();
    }
    if (tid == 0) {
        float relx = redx[0] + bpos[0];
        float rely = redy[0] + bpos[1];
        pos[row * 2 + 0] += relx;
        pos[row * 2 + 1] += rely;
        out_traj[row * 2 + 0] = relx;
        out_traj[row * 2 + 1] = rely;
        float dx = relx - gt[row * 2 + 0], dy = rely - gt[row * 2 + 1];
        lossp[row] = dx * dx + dy * dy;
    }
}

// ---------------------------------------------------------------------------
// Fused pool-net: one block per (group g, agent i). 8 waves, WMMA f16.
//  feats(32x192) -> x1=relu(feats@W1^T+b1)(32x512, LDS f16)
//  -> x2=relu(x1@W2^T+b2) fused with max over the 32 neighbor rows
//  -> dh[row][128 + n]  (f16)
// ---------------------------------------------------------------------------
__global__ __launch_bounds__(256) void pool_kernel(
    const float* __restrict__ h,     // [512][128] (post-LSTM)
    const float* __restrict__ pos,   // [512][2]
    const float* __restrict__ Wp,    // [64][2]
    const float* __restrict__ bp,    // [64]
    const half_t* __restrict__ W1h,  // [512][192]
    const float* __restrict__ b1,    // [512]
    const half_t* __restrict__ W2h,  // [1024][512]
    const float* __restrict__ b2,    // [1024]
    half_t* __restrict__ dh)         // [512][1152]
{
    __shared__ __align__(16) half_t feats[GSZ * PFE_STRIDE];
    __shared__ __align__(16) half_t x1s[GSZ * X1_STRIDE];

    const int blk = blockIdx.x;          // g*32 + i
    const int g = blk >> 5, i = blk & 31;
    const int tid = threadIdx.x;
    const int lane = tid & 31, wave = tid >> 5;
    const int half_ = lane >> 4, l15 = lane & 15;
    const int rowbase = g * GSZ;

    const float pix = pos[(rowbase + i) * 2 + 0];
    const float piy = pos[(rowbase + i) * 2 + 1];

    // feats[j][0:64] = relu-free emb of (pos_j - pos_i);  feats[j][64:192] = h_j
    for (int idx = tid; idx < GSZ * ED; idx += 256) {
        int j = idx >> 6, e = idx & 63;
        float rx = pos[(rowbase + j) * 2 + 0] - pix;
        float ry = pos[(rowbase + j) * 2 + 1] - piy;
        feats[j * PFE_STRIDE + e] = (half_t)(rx * Wp[e * 2 + 0] + ry * Wp[e * 2 + 1] + bp[e]);
    }
    for (int idx = tid; idx < GSZ * HD; idx += 256) {
        int j = idx >> 7, e = idx & 127;
        feats[j * PFE_STRIDE + ED + e] = (half_t)h[(rowbase + j) * HD + e];
    }
    __syncthreads();

    // GEMM1: 2 m-tiles x 32 n-tiles, K=192 (6 wmma per tile)
    for (int t = wave; t < 64; t += 8) {
        const int mt = t & 1, nt = t >> 1;
        v8f acc = {};
        const half_t* arow = &feats[(mt * 16 + l15) * PFE_STRIDE];
        const half_t* brow = W1h + (size_t)(nt * 16 + l15) * KFE;
#pragma unroll
        for (int kk = 0; kk < KFE; kk += 32) {
            v16h a = cat8(*(const v8h*)(arow + kk + half_ * 8),
                          *(const v8h*)(arow + kk + 16 + half_ * 8));
            v16h b = cat8(*(const v8h*)(brow + kk + half_ * 8),
                          *(const v8h*)(brow + kk + 16 + half_ * 8));
            acc = wmma_f16(a, b, acc);
        }
        const int n = nt * 16 + l15;
        const float bias = b1[n];
#pragma unroll
        for (int r = 0; r < 8; r++) {
            int m = mt * 16 + r + half_ * 8;
            float v = acc[r] + bias;
            x1s[m * X1_STRIDE + n] = (half_t)(v > 0.f ? v : 0.f);
        }
    }
    __syncthreads();

    // GEMM2 fused with max over 32 neighbor rows. 64 n-tiles, K=512 (16 wmma x2)
    for (int nt = wave; nt < 64; nt += 8) {
        v8f acc0 = {}, acc1 = {};
        const half_t* a0row = &x1s[l15 * X1_STRIDE];
        const half_t* a1row = &x1s[(16 + l15) * X1_STRIDE];
        const half_t* brow = W2h + (size_t)(nt * 16 + l15) * PRE;
#pragma unroll 4
        for (int kk = 0; kk < PRE; kk += 32) {
            v16h a0 = cat8(*(const v8h*)(a0row + kk + half_ * 8),
                           *(const v8h*)(a0row + kk + 16 + half_ * 8));
            v16h a1 = cat8(*(const v8h*)(a1row + kk + half_ * 8),
                           *(const v8h*)(a1row + kk + 16 + half_ * 8));
            v16h b = cat8(*(const v8h*)(brow + kk + half_ * 8),
                          *(const v8h*)(brow + kk + 16 + half_ * 8));
            acc0 = wmma_f16(a0, b, acc0);
            acc1 = wmma_f16(a1, b, acc1);
        }
        float m = acc0[0];
#pragma unroll
        for (int r = 1; r < 8; r++) m = fmaxf(m, acc0[r]);
#pragma unroll
        for (int r = 0; r < 8; r++) m = fmaxf(m, acc1[r]);
        m = fmaxf(m, __shfl_xor(m, 16, 32));      // combine both row-halves
        const int n = nt * 16 + l15;
        m = fmaxf(m + b2[n], 0.f);                // bias uniform over rows -> after max
        if (lane < 16)
            dh[(size_t)(rowbase + i) * KDH + HD + n] = (half_t)m;
    }
}

// ---------------------------------------------------------------------------
// Decoder MLP: h = relu(relu(dh @ Wm1^T + bm1) @ Wm2^T + bm2). 16 rows / block.
// ---------------------------------------------------------------------------
__global__ __launch_bounds__(256) void decoder_kernel(
    const half_t* __restrict__ dh,    // [512][1152]
    const half_t* __restrict__ Wm1h,  // [1024][1152]
    const float* __restrict__ bm1,    // [1024]
    const half_t* __restrict__ Wm2h,  // [128][1024]
    const float* __restrict__ bm2,    // [128]
    float* __restrict__ h)            // [512][128]
{
    __shared__ __align__(16) half_t mids[16 * MID_STRIDE];
    const int rowbase = blockIdx.x * 16;
    const int tid = threadIdx.x;
    const int lane = tid & 31, wave = tid >> 5;
    const int half_ = lane >> 4, l15 = lane & 15;

    // GEMM1: 1 m-tile x 64 n-tiles, K=1152 (36 wmma per tile)
    const half_t* arow = dh + (size_t)(rowbase + l15) * KDH;
    for (int nt = wave; nt < 64; nt += 8) {
        v8f acc = {};
        const half_t* brow = Wm1h + (size_t)(nt * 16 + l15) * KDH;
        for (int kk = 0; kk < KDH; kk += 32) {
            v16h a = cat8(*(const v8h*)(arow + kk + half_ * 8),
                          *(const v8h*)(arow + kk + 16 + half_ * 8));
            v16h b = cat8(*(const v8h*)(brow + kk + half_ * 8),
                          *(const v8h*)(brow + kk + 16 + half_ * 8));
            acc = wmma_f16(a, b, acc);
        }
        const int n = nt * 16 + l15;
        const float bias = bm1[n];
#pragma unroll
        for (int r = 0; r < 8; r++) {
            int m = r + half_ * 8;
            float v = acc[r] + bias;
            mids[m * MID_STRIDE + n] = (half_t)(v > 0.f ? v : 0.f);
        }
    }
    __syncthreads();

    // GEMM2: 1 m-tile x 8 n-tiles, K=1024 (32 wmma per tile); 1 tile per wave
    {
        const int nt = wave;
        v8f acc = {};
        const half_t* a2row = &mids[l15 * MID_STRIDE];
        const half_t* brow = Wm2h + (size_t)(nt * 16 + l15) * MLPD;
#pragma unroll 4
        for (int kk = 0; kk < MLPD; kk += 32) {
            v16h a = cat8(*(const v8h*)(a2row + kk + half_ * 8),
                          *(const v8h*)(a2row + kk + 16 + half_ * 8));
            v16h b = cat8(*(const v8h*)(brow + kk + half_ * 8),
                          *(const v8h*)(brow + kk + 16 + half_ * 8));
            acc = wmma_f16(a, b, acc);
        }
        const int n = nt * 16 + l15;
        const float bias = bm2[n];
#pragma unroll
        for (int r = 0; r < 8; r++) {
            int m = r + half_ * 8;
            h[(size_t)(rowbase + m) * HD + n] = fmaxf(acc[r] + bias, 0.f);
        }
    }
}

// ---------------------------------------------------------------------------
// Deterministic loss reduction: loss = sum(lossp) / 1024
// ---------------------------------------------------------------------------
__global__ __launch_bounds__(256) void loss_kernel(const float* __restrict__ lossp,
                                                   float* __restrict__ out) {
    __shared__ float red[256];
    const int tid = threadIdx.x;
    float s = 0.f;
    for (int i = tid; i < SEQ * BATCH; i += 256) s += lossp[i];
    red[tid] = s;
    __syncthreads();
    for (int st = 128; st > 0; st >>= 1) {
        if (tid < st) red[tid] += red[tid + st];
        __syncthreads();
    }
    if (tid == 0) out[0] = red[0] * (1.0f / 1024.0f);
}

// ---------------------------------------------------------------------------
extern "C" void kernel_launch(void* const* d_in, const int* in_sizes, int n_in,
                              void* d_out, int out_size, void* d_ws, size_t ws_size,
                              hipStream_t stream) {
    const float* last_pos     = (const float*)d_in[0];
    const float* last_pos_rel = (const float*)d_in[1];
    const float* hh           = (const float*)d_in[2];
    const float* ch           = (const float*)d_in[3];
    const float* ptr_rel      = (const float*)d_in[4];
    // d_in[5] seq_start_end: fixed contiguous groups of 32, baked in as constants
    const float* W_ih = (const float*)d_in[6];
    const float* W_hh = (const float*)d_in[7];
    const float* b_ih = (const float*)d_in[8];
    const float* b_hh = (const float*)d_in[9];
    const float* Wse  = (const float*)d_in[10];
    const float* bse  = (const float*)d_in[11];
    const float* Wpos = (const float*)d_in[12];
    const float* bpos = (const float*)d_in[13];
    const float* Wp   = (const float*)d_in[14];
    const float* bp   = (const float*)d_in[15];
    const float* W1   = (const float*)d_in[16];
    const float* b1   = (const float*)d_in[17];
    const float* W2   = (const float*)d_in[18];
    const float* b2   = (const float*)d_in[19];
    const float* Wm1  = (const float*)d_in[20];
    const float* bm1  = (const float*)d_in[21];
    const float* Wm2  = (const float*)d_in[22];
    const float* bm2  = (const float*)d_in[23];

    char* ws = (char*)d_ws;
    float*  dec_in_all = (float*)ws;  ws += (size_t)SEQ * BATCH * ED * 4;   // 1.5 MB
    float*  h          = (float*)ws;  ws += (size_t)BATCH * HD * 4;         // 256 KB
    float*  c          = (float*)ws;  ws += (size_t)BATCH * HD * 4;
    float*  pos        = (float*)ws;  ws += (size_t)BATCH * 2 * 4;
    half_t* dh         = (half_t*)ws; ws += (size_t)BATCH * KDH * 2;        // 1.13 MB
    half_t* W1h        = (half_t*)ws; ws += (size_t)PRE * KFE * 2;
    half_t* W2h        = (half_t*)ws; ws += (size_t)BOT * PRE * 2;
    half_t* Wm1h       = (half_t*)ws; ws += (size_t)MLPD * KDH * 2;
    half_t* Wm2h       = (half_t*)ws; ws += (size_t)HD * MLPD * 2;
    float*  lossp      = (float*)ws;  ws += (size_t)SEQ * BATCH * 4;

    float* out_traj = (float*)d_out;              // [12][512][2]
    float* out_loss = (float*)d_out + SEQ * BATCH * 2;

    // one-time setup
    cvt_f32_to_f16<<<(PRE * KFE + 255) / 256, 256, 0, stream>>>(W1, W1h, PRE * KFE);
    cvt_f32_to_f16<<<(BOT * PRE + 255) / 256, 256, 0, stream>>>(W2, W2h, BOT * PRE);
    cvt_f32_to_f16<<<(MLPD * KDH + 255) / 256, 256, 0, stream>>>(Wm1, Wm1h, MLPD * KDH);
    cvt_f32_to_f16<<<(HD * MLPD + 255) / 256, 256, 0, stream>>>(Wm2, Wm2h, HD * MLPD);
    init_state<<<(BATCH * HD + 255) / 256, 256, 0, stream>>>(hh, ch, last_pos, h, c, pos);
    decin_kernel<<<(SEQ * BATCH * ED + 255) / 256, 256, 0, stream>>>(
        last_pos_rel, ptr_rel, Wse, bse, dec_in_all);

    for (int t = 0; t < SEQ; t++) {
        lstm_kernel<<<BATCH, 128, 0, stream>>>(
            dec_in_all + (size_t)t * BATCH * ED,
            W_ih, W_hh, b_ih, b_hh, Wpos, bpos,
            ptr_rel + (size_t)t * BATCH * 2,
            h, c, pos, dh,
            out_traj + (size_t)t * BATCH * 2,
            lossp + (size_t)t * BATCH);
        pool_kernel<<<NG * GSZ, 256, 0, stream>>>(h, pos, Wp, bp, W1h, b1, W2h, b2, dh);
        decoder_kernel<<<BATCH / 16, 256, 0, stream>>>(dh, Wm1h, bm1, Wm2h, bm2, h);
    }
    loss_kernel<<<1, 256, 0, stream>>>(lossp, out_loss);
}
